// TransformerModelWithHook_75316546503191
// MI455X (gfx1250) — compile-verified, matrix operand order checked by reading, not run on hardware
//
#include <hip/hip_runtime.h>

// ---------------------------------------------------------------------------
// Types for WMMA bf16 path (gfx1250, wave32)
// ---------------------------------------------------------------------------
typedef __bf16 v16bf __attribute__((ext_vector_type(16)));
typedef float  v8f   __attribute__((ext_vector_type(8)));
typedef unsigned int u32x4 __attribute__((ext_vector_type(4)));
typedef unsigned short ushort_t;

union Frag {
    v16bf v;
    u32x4 q[2];
    ushort_t u[16];
};

static constexpr int B_ = 8, S_ = 1024, D_ = 512, H_ = 8, DH_ = 64;
static constexpr int FF_ = 2048, L_ = 4, OUT_ = 6;
static constexpr int M_ = B_ * S_;                 // 8192 rows of activations
static constexpr size_t WL_ = 3145728;             // bf16 elems per layer's weights

__device__ __forceinline__ ushort_t f2bf(float f) {
    unsigned u = __float_as_uint(f);
    unsigned r = (u + 0x7FFFu + ((u >> 16) & 1u)) >> 16;   // round-nearest-even
    return (ushort_t)r;
}

__device__ __forceinline__ v8f wmma_bf16(const Frag& a, const Frag& b, v8f c) {
    return __builtin_amdgcn_wmma_f32_16x16x32_bf16(false, a.v, false, b.v,
                                                   (short)0, c, false, false);
}

// Async global->LDS copy of 16 bytes per lane (GV addressing, ASYNCcnt).
__device__ __forceinline__ void async_copy_b128(unsigned lds_off,
                                                unsigned long long gaddr) {
    asm volatile("global_load_async_to_lds_b128 %0, %1, off"
                 :: "v"(lds_off), "v"(gaddr) : "memory");
}
__device__ __forceinline__ void wait_asynccnt0() {
    asm volatile("s_wait_asynccnt 0x0" ::: "memory");
}

// DPP16 rotate-based reductions within each 16-lane row (wave32: two halves).
// row_ror:n -> dpp_ctrl = 0x120 + n.  Result broadcast to all 16 lanes of half.
#define DPP_ROR_F32(x, n)                                                     \
    __int_as_float(__builtin_amdgcn_update_dpp(                               \
        0, __float_as_int(x), 0x120 + (n), 0xF, 0xF, true))

__device__ __forceinline__ float rowmax16(float x) {
    x = fmaxf(x, DPP_ROR_F32(x, 8));
    x = fmaxf(x, DPP_ROR_F32(x, 4));
    x = fmaxf(x, DPP_ROR_F32(x, 2));
    x = fmaxf(x, DPP_ROR_F32(x, 1));
    return x;
}
__device__ __forceinline__ float rowsum16(float x) {
    x += DPP_ROR_F32(x, 8);
    x += DPP_ROR_F32(x, 4);
    x += DPP_ROR_F32(x, 2);
    x += DPP_ROR_F32(x, 1);
    return x;
}

// ---------------------------------------------------------------------------
// Weight convert + transpose: src f32 [K,N] -> dst bf16 [N,K]
// ---------------------------------------------------------------------------
__global__ void wtrans(const float* __restrict__ src, ushort_t* __restrict__ dst,
                       int K, int N) {
    int idx = blockIdx.x * 256 + threadIdx.x;
    if (idx >= K * N) return;
    int n = idx / K, kk = idx % K;
    dst[idx] = f2bf(src[(size_t)kk * N + n]);
}

// v bf16 [B,S,D] -> vt bf16 [B,H,DH,S]
__global__ void vtrans(const ushort_t* __restrict__ v, ushort_t* __restrict__ vt) {
    int idx = blockIdx.x * 256 + threadIdx.x;
    if (idx >= M_ * D_) return;
    int s = idx & (S_ - 1);
    int rest = idx >> 10;
    int d = rest & (DH_ - 1);
    int bh = rest >> 6;                 // b*H + h
    int b = bh >> 3, h = bh & 7;
    vt[idx] = v[(size_t)(b * S_ + s) * D_ + h * DH_ + d];
}

// ---------------------------------------------------------------------------
// Embedding: x @ Wproj + bproj + PE, then LayerNorm.  One block per (b,s).
// ---------------------------------------------------------------------------
__global__ void embed_ln(const float* __restrict__ x, const float* __restrict__ Wp,
                         const float* __restrict__ bp, const float* __restrict__ g,
                         const float* __restrict__ beta,
                         float* __restrict__ hOut, ushort_t* __restrict__ hOutB,
                         float* __restrict__ mask) {
    int bs = blockIdx.x;
    int s = bs & (S_ - 1);
    const float* xr = x + (size_t)bs * 3;
    float x0 = xr[0], x1 = xr[1], x2 = xr[2];
    if (threadIdx.x == 0)
        mask[bs] = (fabsf(x0) + fabsf(x1) + fabsf(x2) == 0.0f) ? 1.0f : 0.0f;

    __shared__ float red[256];
    float v[2];
    float pos = (float)s;
    const float c = 9.210340371976184f / 512.0f;   // ln(10000)/D
#pragma unroll
    for (int i = 0; i < 2; ++i) {
        int d = threadIdx.x + i * 256;
        float proj = x0 * Wp[d] + x1 * Wp[D_ + d] + x2 * Wp[2 * D_ + d] + bp[d];
        float freq = __expf(-(float)(d & ~1) * c);
        float ang = pos * freq;
        float pe = (d & 1) ? __cosf(ang) : __sinf(ang);
        v[i] = proj + pe;
    }
    red[threadIdx.x] = v[0] + v[1];
    __syncthreads();
    for (int t = 128; t > 0; t >>= 1) {
        if (threadIdx.x < t) red[threadIdx.x] += red[threadIdx.x + t];
        __syncthreads();
    }
    float mu = red[0] * (1.0f / 512.0f);
    __syncthreads();
    float d0 = v[0] - mu, d1 = v[1] - mu;
    red[threadIdx.x] = d0 * d0 + d1 * d1;
    __syncthreads();
    for (int t = 128; t > 0; t >>= 1) {
        if (threadIdx.x < t) red[threadIdx.x] += red[threadIdx.x + t];
        __syncthreads();
    }
    float rstd = rsqrtf(red[0] * (1.0f / 512.0f) + 1e-5f);
#pragma unroll
    for (int i = 0; i < 2; ++i) {
        int d = threadIdx.x + i * 256;
        float o = (v[i] - mu) * rstd * g[d] + beta[d];
        hOut[(size_t)bs * D_ + d] = o;
        hOutB[(size_t)bs * D_ + d] = f2bf(o);
    }
}

// ---------------------------------------------------------------------------
// (h + res) -> LayerNorm -> f32 (optional) + bf16 (optional). Block per row.
// ---------------------------------------------------------------------------
__global__ void residual_ln(const float* __restrict__ hin, const float* __restrict__ res,
                            const float* __restrict__ g, const float* __restrict__ beta,
                            float* __restrict__ hOut, ushort_t* __restrict__ hOutB) {
    int bs = blockIdx.x;
    __shared__ float red[256];
    float v[2];
#pragma unroll
    for (int i = 0; i < 2; ++i) {
        int d = threadIdx.x + i * 256;
        size_t idx = (size_t)bs * D_ + d;
        float a = hin[idx];
        if (res) a += res[idx];
        v[i] = a;
    }
    red[threadIdx.x] = v[0] + v[1];
    __syncthreads();
    for (int t = 128; t > 0; t >>= 1) {
        if (threadIdx.x < t) red[threadIdx.x] += red[threadIdx.x + t];
        __syncthreads();
    }
    float mu = red[0] * (1.0f / 512.0f);
    __syncthreads();
    float d0 = v[0] - mu, d1 = v[1] - mu;
    red[threadIdx.x] = d0 * d0 + d1 * d1;
    __syncthreads();
    for (int t = 128; t > 0; t >>= 1) {
        if (threadIdx.x < t) red[threadIdx.x] += red[threadIdx.x + t];
        __syncthreads();
    }
    float rstd = rsqrtf(red[0] * (1.0f / 512.0f) + 1e-5f);
#pragma unroll
    for (int i = 0; i < 2; ++i) {
        int d = threadIdx.x + i * 256;
        float o = (v[i] - mu) * rstd * g[d] + beta[d];
        size_t idx = (size_t)bs * D_ + d;
        if (hOut)  hOut[idx]  = o;
        if (hOutB) hOutB[idx] = f2bf(o);
    }
}

// ---------------------------------------------------------------------------
// GEMM: C[M,N] = A[M,K] (bf16) x Bt[N,K]^T (bf16), f32 accumulate via WMMA.
// Block tile 128x64, 8 waves (4 in M x 2 in N), each wave 2x2 16x16 tiles.
// A/B k-slices double-buffered in LDS via async global->LDS DMA (ASYNCcnt):
//   iter i: wait(tile i) -> barrier -> issue tile i+1 -> compute tile i.
// ---------------------------------------------------------------------------
template <bool SILU>
__global__ void gemm_bf16(const ushort_t* __restrict__ A, const ushort_t* __restrict__ Bt,
                          float* __restrict__ Cf, ushort_t* __restrict__ Cb,
                          int M, int N, int K) {
    __shared__ alignas(16) ushort_t ldsA[2][128 * 32];   // 2 x 8 KB
    __shared__ alignas(16) ushort_t ldsB[2][64 * 32];    // 2 x 4 KB

    int tid = threadIdx.x;
    int wave = tid >> 5, lane = tid & 31;
    int wm = wave & 3, wn = wave >> 2;
    int half = lane >> 4, l16 = lane & 15;
    int bm = blockIdx.x * 128, bn = blockIdx.y * 64;

    // Per-thread async-copy assignments: 16B per transfer.
    // B tile: 64 rows x 64B  -> 256 transfers (1/thread)
    // A tile: 128 rows x 64B -> 512 transfers (2/thread)
    int rr = tid >> 2;              // 0..63
    int cc = (tid & 3) * 8;         // ushort offset 0,8,16,24
    const ushort_t* gB  = Bt + (size_t)(bn + rr) * K + cc;
    const ushort_t* gA0 = A + (size_t)(bm + rr) * K + cc;
    const ushort_t* gA1 = A + (size_t)(bm + rr + 64) * K + cc;
    unsigned oB[2], oA0[2], oA1[2];
#pragma unroll
    for (int p = 0; p < 2; ++p) {
        oB[p]  = (unsigned)(size_t)&ldsB[p][rr * 32 + cc];
        oA0[p] = (unsigned)(size_t)&ldsA[p][rr * 32 + cc];
        oA1[p] = (unsigned)(size_t)&ldsA[p][(rr + 64) * 32 + cc];
    }

    // Fragment read pointers (contiguous 16B chunks in LDS).
    const ushort_t* fA = &ldsA[0][(wm * 32 + l16) * 32 + half * 8];
    const ushort_t* fB = &ldsB[0][(wn * 32 + l16) * 32 + half * 8];
    const int strideA = 128 * 32;   // ushorts between LDS A buffers
    const int strideB = 64 * 32;

    // Prologue: DMA first k-slice into buffer 0.
    async_copy_b128(oB[0],  (unsigned long long)gB);
    async_copy_b128(oA0[0], (unsigned long long)gA0);
    async_copy_b128(oA1[0], (unsigned long long)gA1);

    v8f acc[2][2] = {};
    int cur = 0;
    for (int k0 = 0; k0 < K; k0 += 32) {
        wait_asynccnt0();        // tile k0 landed (tile k0+32 not yet issued)
        __syncthreads();         // all waves: data visible; prev compute done
        if (k0 + 32 < K) {
            int nxt = cur ^ 1;
            async_copy_b128(oB[nxt],  (unsigned long long)(gB  + k0 + 32));
            async_copy_b128(oA0[nxt], (unsigned long long)(gA0 + k0 + 32));
            async_copy_b128(oA1[nxt], (unsigned long long)(gA1 + k0 + 32));
        }

        Frag a[2], b[2];
#pragma unroll
        for (int t = 0; t < 2; ++t) {
            const ushort_t* pa = fA + cur * strideA + t * 16 * 32;
            a[t].q[0] = *(const u32x4*)pa;
            a[t].q[1] = *(const u32x4*)(pa + 16);
            const ushort_t* pb = fB + cur * strideB + t * 16 * 32;
            b[t].q[0] = *(const u32x4*)pb;
            b[t].q[1] = *(const u32x4*)(pb + 16);
        }
#pragma unroll
        for (int i = 0; i < 2; ++i)
#pragma unroll
            for (int j = 0; j < 2; ++j)
                acc[i][j] = wmma_bf16(a[i], b[j], acc[i][j]);
        cur ^= 1;
    }

    int rowA0 = bm + wm * 32;
    int colB0 = bn + wn * 32;
#pragma unroll
    for (int i = 0; i < 2; ++i)
#pragma unroll
        for (int j = 0; j < 2; ++j)
#pragma unroll
            for (int r = 0; r < 8; ++r) {
                int row = rowA0 + i * 16 + half * 8 + r;
                int col = colB0 + j * 16 + l16;
                float v = acc[i][j][r];
                if (SILU) v = v / (1.0f + __expf(-v));
                size_t idx = (size_t)row * N + col;
                if (Cf) Cf[idx] = v;
                if (Cb) Cb[idx] = f2bf(v);
            }
}

// ---------------------------------------------------------------------------
// Flash-style attention. Wave handles one 16-query tile of one (b,h).
// Block = 4 waves. q,k: bf16 [B,S,D]; vt: bf16 [B,H,DH,S]; out: bf16 [B,S,D].
// ---------------------------------------------------------------------------
__global__ void attention(const ushort_t* __restrict__ q, const ushort_t* __restrict__ k,
                          const ushort_t* __restrict__ vt, const float* __restrict__ mask,
                          ushort_t* __restrict__ out) {
    int wave = threadIdx.x >> 5, lane = threadIdx.x & 31;
    int qt = blockIdx.x * 4 + wave;
    int b = qt / (H_ * (S_ / 16));
    int rem = qt % (H_ * (S_ / 16));
    int h = rem / (S_ / 16);
    int q0 = (rem % (S_ / 16)) * 16;
    int half = lane >> 4, l16 = lane & 15, koff = half * 8;

    __shared__ alignas(16) ushort_t pbuf[4][16][32];
    ushort_t(*pw)[32] = pbuf[wave];

    // Q A-fragments (two 32-wide dh chunks), loaded once.
    Frag aq[2];
    const ushort_t* Qb = q + (size_t)(b * S_ + q0 + l16) * D_ + h * DH_;
#pragma unroll
    for (int dc = 0; dc < 2; ++dc) {
        const ushort_t* p = Qb + dc * 32 + koff;
        aq[dc].q[0] = *(const u32x4*)p;
        aq[dc].q[1] = *(const u32x4*)(p + 16);
    }
    const ushort_t* Kb = k + (size_t)b * S_ * D_ + h * DH_;
    const ushort_t* Vb = vt + (size_t)(b * H_ + h) * DH_ * S_;
    const float* mb = mask + (size_t)b * S_;

    float mrow[8], lrow[8];
    v8f oacc[4] = {};
#pragma unroll
    for (int r = 0; r < 8; ++r) { mrow[r] = -1e30f; lrow[r] = 0.0f; }
    const float scale = 0.125f;   // 1/sqrt(64)

    for (int key0 = 0; key0 < S_; key0 += 32) {
        v8f s[2] = {};
#pragma unroll
        for (int nt = 0; nt < 2; ++nt) {
            int key = key0 + nt * 16 + l16;
            Frag bk[2];
#pragma unroll
            for (int dc = 0; dc < 2; ++dc) {
                const ushort_t* p = Kb + (size_t)key * D_ + dc * 32 + koff;
                bk[dc].q[0] = *(const u32x4*)p;
                bk[dc].q[1] = *(const u32x4*)(p + 16);
            }
            s[nt] = wmma_bf16(aq[0], bk[0], s[nt]);
            s[nt] = wmma_bf16(aq[1], bk[1], s[nt]);
            float mpen = mb[key] * -1e30f;
#pragma unroll
            for (int r = 0; r < 8; ++r) s[nt][r] = s[nt][r] * scale + mpen;
        }
        // online softmax per query row (VGPR r; lane halves hold rows r / r+8)
        // DPP rotate reductions: pure VALU, no LDS round-trips.
#pragma unroll
        for (int r = 0; r < 8; ++r) {
            float mx = rowmax16(fmaxf(s[0][r], s[1][r]));
            float mnew = fmaxf(mrow[r], mx);
            float p0 = __expf(s[0][r] - mnew);
            float p1 = __expf(s[1][r] - mnew);
            float sum = rowsum16(p0 + p1);
            float alpha = __expf(mrow[r] - mnew);
            lrow[r] = lrow[r] * alpha + sum;
            mrow[r] = mnew;
#pragma unroll
            for (int t = 0; t < 4; ++t) oacc[t][r] *= alpha;
            int row = half * 8 + r;
            pw[row][l16] = f2bf(p0);
            pw[row][16 + l16] = f2bf(p1);
        }
        // restripe P into A-operand layout (LDS ops from same wave are in order)
        Frag pf;
        {
            const ushort_t* p = &pw[l16][koff];
            pf.q[0] = *(const u32x4*)p;
            pf.q[1] = *(const u32x4*)(p + 16);
        }
#pragma unroll
        for (int t = 0; t < 4; ++t) {
            Frag bv;
            const ushort_t* p = Vb + (size_t)(t * 16 + l16) * S_ + key0 + koff;
            bv.q[0] = *(const u32x4*)p;
            bv.q[1] = *(const u32x4*)(p + 16);
            oacc[t] = wmma_bf16(pf, bv, oacc[t]);
        }
    }
    ushort_t* ob = out + (size_t)(b * S_ + q0) * D_ + h * DH_;
#pragma unroll
    for (int t = 0; t < 4; ++t)
#pragma unroll
        for (int r = 0; r < 8; ++r) {
            float v = oacc[t][r] / lrow[r];
            ob[(size_t)(half * 8 + r) * D_ + t * 16 + l16] = f2bf(v);
        }
}

// ---------------------------------------------------------------------------
// Masked mean pool + output projection (tiny, scalar f32).
// ---------------------------------------------------------------------------
__global__ void pool(const float* __restrict__ hf, const float* __restrict__ mask,
                     float* __restrict__ pooled) {
    int b = blockIdx.x;
    int d = threadIdx.x;
    float acc = 0.0f, cnt = 0.0f;
    for (int s = 0; s < S_; ++s) {
        float np = 1.0f - mask[(size_t)b * S_ + s];
        acc += hf[((size_t)b * S_ + s) * D_ + d] * np;
        cnt += np;
    }
    pooled[(size_t)b * D_ + d] = acc / (cnt + 1e-8f);
}

__global__ void outproj(const float* __restrict__ pooled, const float* __restrict__ Wout,
                        const float* __restrict__ bout, float* __restrict__ out) {
    int i = threadIdx.x;
    if (i >= B_ * OUT_) return;
    int b = i / OUT_, o = i % OUT_;
    float acc = bout[o];
    for (int d = 0; d < D_; ++d)
        acc += pooled[(size_t)b * D_ + d] * Wout[(size_t)d * OUT_ + o];
    out[i] = acc;
}

// ---------------------------------------------------------------------------
// Host launcher
// ---------------------------------------------------------------------------
extern "C" void kernel_launch(void* const* d_in, const int* in_sizes, int n_in,
                              void* d_out, int out_size, void* d_ws, size_t ws_size,
                              hipStream_t stream) {
    (void)in_sizes; (void)n_in; (void)out_size; (void)ws_size;
    const float* x     = (const float*)d_in[0];
    const float* Wproj = (const float*)d_in[1];
    const float* bproj = (const float*)d_in[2];
    const float* gp    = (const float*)d_in[3];
    const float* bp    = (const float*)d_in[4];
    const float* Wq    = (const float*)d_in[5];
    const float* Wk    = (const float*)d_in[6];
    const float* Wv    = (const float*)d_in[7];
    const float* Wo    = (const float*)d_in[8];
    const float* g1    = (const float*)d_in[9];
    const float* b1    = (const float*)d_in[10];
    const float* W1    = (const float*)d_in[11];
    const float* W2    = (const float*)d_in[12];
    const float* g2    = (const float*)d_in[13];
    const float* b2    = (const float*)d_in[14];
    const float* gf    = (const float*)d_in[15];
    const float* bfin  = (const float*)d_in[16];
    const float* Wout  = (const float*)d_in[17];
    const float* bout  = (const float*)d_in[18];
    float* out = (float*)d_out;

    char* ws = (char*)d_ws;
    size_t off = 0;
    auto alloc = [&](size_t bytes) -> void* {
        void* p = ws + off;
        off += (bytes + 255) & ~(size_t)255;
        return p;
    };
    float*    h_f    = (float*)alloc((size_t)M_ * D_ * 4);
    ushort_t* h_b    = (ushort_t*)alloc((size_t)M_ * D_ * 2);
    ushort_t* q_b    = (ushort_t*)alloc((size_t)M_ * D_ * 2);
    ushort_t* k_b    = (ushort_t*)alloc((size_t)M_ * D_ * 2);
    ushort_t* v_b    = (ushort_t*)alloc((size_t)M_ * D_ * 2);
    ushort_t* vt_b   = (ushort_t*)alloc((size_t)M_ * D_ * 2);
    ushort_t* ff1_b  = q_b;  // alias: q+k+v+vt = 32MB = M*FF*2 bytes, contiguous
    ushort_t* at_b   = (ushort_t*)alloc((size_t)M_ * D_ * 2);
    float*    o_f    = (float*)alloc((size_t)M_ * D_ * 4);
    float*    maskp  = (float*)alloc((size_t)B_ * S_ * 4);
    float*    pooled = (float*)alloc((size_t)B_ * D_ * 4);
    ushort_t* wbase  = (ushort_t*)alloc(WL_ * L_ * 2);

    // ---- weight conversion: f32 [K,N] -> bf16 [N,K] ----
    for (int l = 0; l < L_; ++l) {
        ushort_t* wl = wbase + WL_ * l;
        int gDD = (D_ * D_ + 255) / 256;
        wtrans<<<gDD, 256, 0, stream>>>(Wq + (size_t)l * D_ * D_, wl + 0,      D_, D_);
        wtrans<<<gDD, 256, 0, stream>>>(Wk + (size_t)l * D_ * D_, wl + 262144, D_, D_);
        wtrans<<<gDD, 256, 0, stream>>>(Wv + (size_t)l * D_ * D_, wl + 524288, D_, D_);
        wtrans<<<gDD, 256, 0, stream>>>(Wo + (size_t)l * D_ * D_, wl + 786432, D_, D_);
        int gDF = (D_ * FF_ + 255) / 256;
        wtrans<<<gDF, 256, 0, stream>>>(W1 + (size_t)l * D_ * FF_, wl + 1048576, D_, FF_);
        wtrans<<<gDF, 256, 0, stream>>>(W2 + (size_t)l * FF_ * D_, wl + 2097152, FF_, D_);
    }

    // ---- embedding + first LN ----
    embed_ln<<<M_, 256, 0, stream>>>(x, Wproj, bproj, gp, bp, h_f, h_b, maskp);

    dim3 gD(M_ / 128, D_ / 64);
    dim3 gF(M_ / 128, FF_ / 64);
    for (int l = 0; l < L_; ++l) {
        ushort_t* wq = wbase + WL_ * l;
        ushort_t* wk = wq + 262144;
        ushort_t* wv = wq + 524288;
        ushort_t* wo = wq + 786432;
        ushort_t* w1 = wq + 1048576;
        ushort_t* w2 = wq + 2097152;

        gemm_bf16<false><<<gD, 256, 0, stream>>>(h_b, wq, nullptr, q_b, M_, D_, D_);
        gemm_bf16<false><<<gD, 256, 0, stream>>>(h_b, wk, nullptr, k_b, M_, D_, D_);
        gemm_bf16<false><<<gD, 256, 0, stream>>>(h_b, wv, nullptr, v_b, M_, D_, D_);
        vtrans<<<(M_ * D_ + 255) / 256, 256, 0, stream>>>(v_b, vt_b);
        attention<<<(B_ * H_ * (S_ / 16)) / 4, 128, 0, stream>>>(q_b, k_b, vt_b, maskp, at_b);
        gemm_bf16<false><<<gD, 256, 0, stream>>>(at_b, wo, o_f, nullptr, M_, D_, D_);
        residual_ln<<<M_, 256, 0, stream>>>(h_f, o_f, g1 + (size_t)l * D_, b1 + (size_t)l * D_, h_f, h_b);
        gemm_bf16<true><<<gF, 256, 0, stream>>>(h_b, w1, nullptr, ff1_b, M_, FF_, D_);
        gemm_bf16<false><<<gD, 256, 0, stream>>>(ff1_b, w2, o_f, nullptr, M_, D_, FF_);
        residual_ln<<<M_, 256, 0, stream>>>(h_f, o_f, g2 + (size_t)l * D_, b2 + (size_t)l * D_, h_f, h_b);
    }

    // ---- final LN -> o_f, masked mean pool, output projection ----
    residual_ln<<<M_, 256, 0, stream>>>(h_f, nullptr, gf, bfin, o_f, nullptr);
    pool<<<B_, 512, 0, stream>>>(o_f, maskp, pooled);
    outproj<<<1, 64, 0, stream>>>(pooled, Wout, bout, out);
}
